// NeighborhoodAttention3D_83184926589046
// MI455X (gfx1250) — matrix operand order repaired and checked by reading, not compiled, not run on hardware
//
#include <hip/hip_runtime.h>
#include <hip/hip_bf16.h>

typedef __attribute__((ext_vector_type(16))) _Float16 v16h;
typedef __attribute__((ext_vector_type(8)))  float    v8f;
typedef __attribute__((ext_vector_type(4)))  unsigned int u32x4;
typedef __attribute__((ext_vector_type(8)))  int i32x8;
typedef __attribute__((ext_vector_type(4)))  int i32x4;

#define BB 2
#define DD 16
#define HH 32
#define WW 32
#define CC 256
#define NH 8
#define HD 32
#define KD 3
#define KK 7
#define NEIGH 147
#define TOKENS (BB*DD*HH*WW)              // 32768
#define HEADVOL (DD*HH*WW*HD)             // 524288 per (b,head)
#define QKV_ELEMS ((size_t)BB*NH*DD*HH*WW*HD)   // 8388608

// ================= TDM: 2-D tile load global(f16) -> LDS =================
// Builds a D# (groups 0/1 per CDNA5 ISA 8.3/8.4) for a 2-D f16 tile and issues
// tensor_load_to_lds (6-arg clang-23 form). tile_w/tile_h/stride in f16 elements.
__device__ __forceinline__ void tdm_load_2d(unsigned lds_off, const void* gsrc,
                                            unsigned tile_w, unsigned tile_h,
                                            unsigned stride) {
    const unsigned long long ga = (unsigned long long)(uintptr_t)gsrc;
    u32x4 g0;
    g0[0] = 1u;                                    // count=1, user descriptor
    g0[1] = lds_off;                               // lds_addr (bytes)
    g0[2] = (unsigned)(ga & 0xFFFFFFFFu);          // global_addr[31:0]
    g0[3] = (unsigned)((ga >> 32) & 0x1FFFFFFu) | (2u << 30);  // addr[56:32] | type=2

    const unsigned td1 = 1u << 20;                 // generous tensor_dim1 (no OOB)
    i32x8 g1;
    g1[0] = (int)(1u << 16);                       // wg_mask=0, data_size=1 (2 bytes)
    g1[1] = (int)((stride & 0xFFFFu) << 16);       // barrier_addr=0 | tensor_dim0[15:0]
    g1[2] = (int)(((stride >> 16) & 0xFFFFu) |     // tensor_dim0[31:16]
                  ((td1 & 0xFFFFu) << 16));        // tensor_dim1[15:0]
    g1[3] = (int)(((td1 >> 16) & 0xFFFFu) |        // tensor_dim1[31:16]
                  ((tile_w & 0xFFFFu) << 16));     // tile_dim0
    g1[4] = (int)(tile_h & 0xFFFFu);               // tile_dim1 | tile_dim2=0
    g1[5] = (int)stride;                           // tensor_dim0_stride[31:0]
    g1[6] = 0;                                     // stride[47:32] | dim1_stride lo
    g1[7] = 0;

    i32x4 z4; z4[0] = 0; z4[1] = 0; z4[2] = 0; z4[3] = 0;
    i32x8 z8;
#pragma unroll
    for (int i = 0; i < 8; ++i) z8[i] = 0;
    __builtin_amdgcn_tensor_load_to_lds(g0, g1, z4, z4, z8, 0);
}

// ================= WMMA fragment loads from LDS (f16) =================
// A tile in LDS: rows x 32 (K) row-major. ISA A layout: lane -> m=lane&15,
// half=lane>>4; vgpr v holds K=(v/4)*16 + half*8 + (v&3)*2 (+1).
__device__ __forceinline__ v16h lds_a_frag(const _Float16* As, int mrow0, int lane) {
    v16h a;
    const _Float16* ap = As + (mrow0 + (lane & 15)) * 32;
    const int hh = lane >> 4;
#pragma unroll
    for (int v = 0; v < 8; ++v) {
        const int kb = ((v >> 2) << 4) + (hh << 3) + ((v & 3) << 1);
        a[2 * v]     = ap[kb];
        a[2 * v + 1] = ap[kb + 1];
    }
    return a;
}
// B tile in LDS: 32 (K) x 64 row-major. ISA B layout: lane -> n=lane&15,
// half=lane>>4; element s holds K=(s/8)*16 + half*8 + (s&7) (same as A).
__device__ __forceinline__ v16h lds_b_frag(const _Float16* Bs, int col0, int lane) {
    v16h b;
    const _Float16* bp = Bs + col0 + (lane & 15);
    const int hh = lane >> 4;
#pragma unroll
    for (int s = 0; s < 16; ++s) {
        const int k = ((s >> 3) << 4) + (hh << 3) + (s & 7);
        b[s] = bp[k * 64];
    }
    return b;
}

// ================= fp32 -> f16 conversion =================
__global__ void f32_to_f16_kernel(const float* __restrict__ src,
                                  _Float16* __restrict__ dst, int n) {
    const int i = (blockIdx.x * 256 + threadIdx.x) * 4;
#pragma unroll
    for (int t = 0; t < 4; ++t)
        if (i + t < n) dst[i + t] = (_Float16)src[i + t];
}

// ================= Kernel 1: QKV GEMM (M=32768,N=768,K=256) =================
// 4 waves / block; block tile 64x64; wave tile 16x64; TDM-staged LDS operands.
__global__ void __launch_bounds__(128)
qkv_gemm_kernel(const _Float16* __restrict__ xh, const _Float16* __restrict__ wh,
                const float* __restrict__ bias,
                float* __restrict__ Qb, float* __restrict__ Kb, float* __restrict__ Vb) {
    __shared__ __align__(32) _Float16 As[64 * 32];
    __shared__ __align__(32) _Float16 Bs[32 * 64];
    const int tid = threadIdx.x;
    const int lane = tid & 31;
    const int wv = tid >> 5;
    const int row0 = blockIdx.x * 64;
    const int col0 = blockIdx.y * 64;

    v8f acc[4] = {};
    for (int k0 = 0; k0 < CC; k0 += 32) {
        if (tid < 32) {                       // wave-uniform: only wave 0 issues DMA
            tdm_load_2d((unsigned)(uintptr_t)As, xh + (size_t)row0 * CC + k0, 32, 64, CC);
            tdm_load_2d((unsigned)(uintptr_t)Bs, wh + (size_t)k0 * (3 * CC) + col0, 64, 32, 3 * CC);
            __builtin_amdgcn_s_wait_tensorcnt(0);
        }
        __syncthreads();
        v16h a = lds_a_frag(As, wv * 16, lane);
#pragma unroll
        for (int nt = 0; nt < 4; ++nt) {
            v16h b = lds_b_frag(Bs, nt * 16, lane);
            acc[nt] = __builtin_amdgcn_wmma_f32_16x16x32_f16(false, a, false, b,
                                                             (short)0, acc[nt], false, false);
        }
        __syncthreads();
    }

    // scatter q(*scale)/k/v into head-major fp32 buffers
    const int mbase = row0 + wv * 16 + ((lane >> 4) << 3);
#pragma unroll
    for (int nt = 0; nt < 4; ++nt) {
        const int col = col0 + nt * 16 + (lane & 15);
        const int g = col >> 8, nHd = (col >> 5) & 7, ch = col & 31;
        const float bv = bias[col];
        const float scl = (g == 0) ? 0.17677669529663687f : 1.0f;
        float* dst = (g == 0) ? Qb : ((g == 1) ? Kb : Vb);
#pragma unroll
        for (int r = 0; r < 8; ++r) {
            const int m = mbase + r;
            const int b_ = m >> 14, d = (m >> 10) & 15, h = (m >> 5) & 31, w_ = m & 31;
            const size_t idx = (size_t)(b_ * NH + nHd) * HEADVOL +
                               (size_t)((d * HH + h) * WW + w_) * HD + ch;
            dst[idx] = (acc[nt][r] + bv) * scl;
        }
    }
}

// ================= Kernel 2: neighborhood attention (fp32, wave32) =================
__global__ void natten_kernel(const float* __restrict__ Qb, const float* __restrict__ Kb,
                              const float* __restrict__ Vb, const float* __restrict__ rpb,
                              float* __restrict__ AO) {
    __shared__ float attn[160];
    const int lane = threadIdx.x;
    const int blk = blockIdx.x;               // B*NH*D*H = 8192
    const int h = blk & 31;
    const int d = (blk >> 5) & 15;
    const int nHd = (blk >> 9) & 7;
    const int b_ = blk >> 12;

    const int sd = min(max(d - 1, 0), DD - KD);
    const int sh = min(max(h - 3, 0), HH - KK);
    const float* rp = rpb + nHd * (2 * KD - 1) * (2 * KK - 1) * (2 * KK - 1);
    const size_t plane = (size_t)(b_ * NH + nHd) * HEADVOL;

    for (int w = 0; w < WW; ++w) {
        const int sw = min(max(w - 3, 0), WW - KK);
        const float* qp = Qb + plane + (size_t)((d * HH + h) * WW + w) * HD;
        float4 q4[8];
#pragma unroll
        for (int t = 0; t < 8; ++t) q4[t] = ((const float4*)qp)[t];

        float l[5];
#pragma unroll
        for (int r = 0; r < 5; ++r) {
            const int j = r * 32 + lane;
            float val = -3.0e38f;
            if (j < NEIGH) {
                const int od = j / 49;
                const int rem = j - od * 49;
                const int oh = rem / 7;
                const int ow = rem - oh * 7;
                const int jd = sd + od, jh = sh + oh, jw = sw + ow;
                const size_t noff = plane + (size_t)((jd * HH + jh) * WW + jw) * HD;
                __builtin_prefetch(Vb + noff, 0, 1);   // warm V for the AV pass
                const float4* kp = (const float4*)(Kb + noff);
                float dot = 0.f;
#pragma unroll
                for (int t = 0; t < 8; ++t) {
                    float4 kv = kp[t];
                    dot += q4[t].x * kv.x + q4[t].y * kv.y +
                           q4[t].z * kv.z + q4[t].w * kv.w;
                }
                const int bd = jd - d + (KD - 1);
                const int bh = jh - h + (KK - 1);
                const int bw = jw - w + (KK - 1);
                val = dot + rp[(bd * 13 + bh) * 13 + bw];
            }
            l[r] = val;
        }

        float mx = l[0];
#pragma unroll
        for (int r = 1; r < 5; ++r) mx = fmaxf(mx, l[r]);
#pragma unroll
        for (int off = 16; off >= 1; off >>= 1) mx = fmaxf(mx, __shfl_xor(mx, off, 32));
        float e[5], s = 0.f;
#pragma unroll
        for (int r = 0; r < 5; ++r) { e[r] = __expf(l[r] - mx); s += e[r]; }
#pragma unroll
        for (int off = 16; off >= 1; off >>= 1) s += __shfl_xor(s, off, 32);
        const float inv = 1.0f / s;
#pragma unroll
        for (int r = 0; r < 5; ++r) attn[r * 32 + lane] = e[r] * inv;
        __syncthreads();

        float acc = 0.f;
        const int c = lane;
#pragma unroll
        for (int od = 0; od < KD; ++od)
#pragma unroll
            for (int oh = 0; oh < KK; ++oh) {
                const int jd = sd + od, jh = sh + oh;
                const float* vrow = Vb + plane + (size_t)((jd * HH + jh) * WW + sw) * HD;
                const int jb = (od * KK + oh) * KK;
#pragma unroll
                for (int ow = 0; ow < KK; ++ow)
                    acc += attn[jb + ow] * vrow[ow * HD + c];
            }

        AO[(size_t)(((b_ * DD + d) * HH + h) * WW + w) * CC + nHd * HD + c] = acc;
        __syncthreads();
    }
}

// ================= Kernel 3: projection GEMM (M=32768,N=256,K=256) =================
__global__ void __launch_bounds__(128)
proj_gemm_kernel(const _Float16* __restrict__ aoh, const _Float16* __restrict__ wh,
                 const float* __restrict__ bias, float* __restrict__ out) {
    __shared__ __align__(32) _Float16 As[64 * 32];
    __shared__ __align__(32) _Float16 Bs[32 * 64];
    const int tid = threadIdx.x;
    const int lane = tid & 31;
    const int wv = tid >> 5;
    const int row0 = blockIdx.x * 64;
    const int col0 = blockIdx.y * 64;

    v8f acc[4] = {};
    for (int k0 = 0; k0 < CC; k0 += 32) {
        if (tid < 32) {
            tdm_load_2d((unsigned)(uintptr_t)As, aoh + (size_t)row0 * CC + k0, 32, 64, CC);
            tdm_load_2d((unsigned)(uintptr_t)Bs, wh + (size_t)k0 * CC + col0, 64, 32, CC);
            __builtin_amdgcn_s_wait_tensorcnt(0);
        }
        __syncthreads();
        v16h a = lds_a_frag(As, wv * 16, lane);
#pragma unroll
        for (int nt = 0; nt < 4; ++nt) {
            v16h b = lds_b_frag(Bs, nt * 16, lane);
            acc[nt] = __builtin_amdgcn_wmma_f32_16x16x32_f16(false, a, false, b,
                                                             (short)0, acc[nt], false, false);
        }
        __syncthreads();
    }

    const int mbase = row0 + wv * 16 + ((lane >> 4) << 3);
#pragma unroll
    for (int nt = 0; nt < 4; ++nt) {
        const int col = col0 + nt * 16 + (lane & 15);
        const float bv = bias[col];
#pragma unroll
        for (int r = 0; r < 8; ++r)
            out[(size_t)(mbase + r) * CC + col] = acc[nt][r] + bv;
    }
}

extern "C" void kernel_launch(void* const* d_in, const int* in_sizes, int n_in,
                              void* d_out, int out_size, void* d_ws, size_t ws_size,
                              hipStream_t stream) {
    const float* x      = (const float*)d_in[0];
    const float* w_qkv  = (const float*)d_in[1];
    const float* b_qkv  = (const float*)d_in[2];
    const float* rpb    = (const float*)d_in[3];
    const float* w_proj = (const float*)d_in[4];
    const float* b_proj = (const float*)d_in[5];
    float* out = (float*)d_out;

    // workspace carve-up
    float* Qb = (float*)d_ws;
    float* Kb = Qb + QKV_ELEMS;
    float* Vb = Kb + QKV_ELEMS;
    float* AO = Vb + QKV_ELEMS;
    _Float16* xh   = (_Float16*)(AO + QKV_ELEMS);
    _Float16* aoh  = xh + QKV_ELEMS;
    _Float16* wqh  = aoh + QKV_ELEMS;              // 256*768
    _Float16* wph  = wqh + (size_t)CC * 3 * CC;    // 256*256

    const int nx  = (int)QKV_ELEMS;                // 8388608
    const int nwq = CC * 3 * CC;                   // 196608
    const int nwp = CC * CC;                       // 65536

    f32_to_f16_kernel<<<(nx  + 1023) / 1024, 256, 0, stream>>>(x, xh, nx);
    f32_to_f16_kernel<<<(nwq + 1023) / 1024, 256, 0, stream>>>(w_qkv, wqh, nwq);
    f32_to_f16_kernel<<<(nwp + 1023) / 1024, 256, 0, stream>>>(w_proj, wph, nwp);

    qkv_gemm_kernel<<<dim3(TOKENS / 64, (3 * CC) / 64), 128, 0, stream>>>(
        xh, wqh, b_qkv, Qb, Kb, Vb);

    natten_kernel<<<BB * NH * DD * HH, 32, 0, stream>>>(Qb, Kb, Vb, rpb, AO);

    f32_to_f16_kernel<<<(nx + 1023) / 1024, 256, 0, stream>>>(AO, aoh, nx);

    proj_gemm_kernel<<<dim3(TOKENS / 64, CC / 64), 128, 0, stream>>>(
        aoh, wph, b_proj, out);
}
